// WTA_9526237462867
// MI455X (gfx1250) — compile-verified, matrix operand order checked by reading, not run on hardware
//
#include <hip/hip_runtime.h>
#include <cstdint>

// ---------------------------------------------------------------------------
// Restormer-style transposed attention over H axis, CDNA5 (gfx1250, wave32).
// WMMA f16->f32 for the attention GEMMs, f32 VALU for the small convs.
// ---------------------------------------------------------------------------

typedef __attribute__((ext_vector_type(16))) _Float16 v16h;
typedef __attribute__((ext_vector_type(8)))  float    v8f;

union AU { uint32_t u[8]; _Float16 h[16]; v16h v; };

constexpr int BN   = 4;      // batch
constexpr int CDIM = 48;     // channels
constexpr int NH   = 3;      // heads
constexpr int HH   = 256;    // height (attention sequence length)
constexpr int WW   = 256;    // width
constexpr int FF   = 4096;   // feature dim per head row = (C/NH)*W = 16*256
constexpr int C3   = 144;    // 3*C
constexpr int HW   = HH * WW;

// ---------------------------------------------------------------------------
// K0: zero the sum-of-squares accumulators (ws is poisoned 0xAA by harness)
// ---------------------------------------------------------------------------
__global__ void k_zero(float* __restrict__ sq, float* __restrict__ sk) {
  int i = blockIdx.x * 256 + threadIdx.x;
  if (i < BN * NH * HH) { sq[i] = 0.f; sk[i] = 0.f; }
}

// ---------------------------------------------------------------------------
// K1: conv1x1 48 -> 144, output f16. LDS tile of 128 pixels x 48 channels.
// ---------------------------------------------------------------------------
__global__ void k_conv1(const float* __restrict__ x, const float* __restrict__ wq,
                        _Float16* __restrict__ pre) {
  const int tile = blockIdx.x & 511;        // 512 tiles of 128 pixels per batch
  const int b    = blockIdx.x >> 9;
  const int p0   = tile * 128;
  __shared__ float xs[48][132];
  for (int i = threadIdx.x; i < 48 * 128; i += 256) {
    int c = i >> 7, p = i & 127;
    xs[c][p] = x[(size_t)(b * CDIM + c) * HW + p0 + p];
  }
  __syncthreads();
  const int p  = threadIdx.x & 127;
  const int og = threadIdx.x >> 7;          // uniform within a wave
  for (int o = og; o < C3; o += 2) {
    float acc = 0.f;
    const float* wrow = wq + o * CDIM;      // scalar (uniform) reads
    #pragma unroll 8
    for (int c = 0; c < CDIM; ++c) acc += wrow[c] * xs[c][p];
    pre[(size_t)(b * C3 + o) * HW + p0 + p] = (_Float16)acc;
  }
}

// ---------------------------------------------------------------------------
// K2: depthwise 3x3 (SAME, zero pad) + reorder into head layout
//     [b, head, h, f = cc*W + w] (f16), plus per-row sum(q^2), sum(k^2).
// One block = one (b, channel, h) row of 256 pixels.
// ---------------------------------------------------------------------------
__global__ void k_dw(const _Float16* __restrict__ pre, const float* __restrict__ wd,
                     _Float16* __restrict__ qh, _Float16* __restrict__ kh,
                     _Float16* __restrict__ vh,
                     float* __restrict__ sq, float* __restrict__ sk) {
  const int h  = blockIdx.x & 255;
  const int ch = (blockIdx.x >> 8) % C3;
  const int b  = blockIdx.x / (256 * C3);
  const int w  = threadIdx.x;
  const int sel = ch / CDIM, rem = ch % CDIM, head = rem >> 4, cc = rem & 15;

  float wdr[9];
  #pragma unroll
  for (int i = 0; i < 9; ++i) wdr[i] = wd[ch * 9 + i];

  float acc = 0.f;
  #pragma unroll
  for (int dy = 0; dy < 3; ++dy) {
    int hy = h + dy - 1;
    if (hy < 0 || hy > 255) continue;
    const _Float16* row = pre + (size_t)((b * C3 + ch) * HH + hy) * WW;
    #pragma unroll
    for (int dx = 0; dx < 3; ++dx) {
      int wx = w + dx - 1;
      if (wx < 0 || wx > 255) continue;
      acc += (float)row[wx] * wdr[dy * 3 + dx];
    }
  }
  const int rowi = (b * NH + head) * HH + h;
  _Float16* dst = (sel == 0) ? qh : (sel == 1) ? kh : vh;
  dst[(size_t)rowi * FF + cc * WW + w] = (_Float16)acc;

  if (sel < 2) {   // sel uniform per block
    __shared__ float red[256];
    red[w] = acc * acc;
    __syncthreads();
    for (int s = 128; s > 0; s >>= 1) {
      if (w < s) red[w] += red[w + s];
      __syncthreads();
    }
    if (w == 0) atomicAdd((sel == 0 ? sq : sk) + rowi, red[0]);
  }
}

// ---------------------------------------------------------------------------
// K3: attention per (b, head, 16-row tile). 256 threads = 8 wave32.
//   Phase 1: S = Q K^T via v_wmma_f32_16x16x32_f16 (raw f16 dots, f32 acc)
//   Phase 2: scale by rq_i*rk_j*temperature, softmax over 256 cols, f16 probs
//   Phase 3: O = P V via WMMA; V staged in LDS pair-interleaved so each lane
//            builds its B operand with 8 ds_load_b32 (not 16 half loads)
// ---------------------------------------------------------------------------
__global__ void k_attn(const _Float16* __restrict__ qh, const _Float16* __restrict__ kh,
                       const _Float16* __restrict__ vh, const float* __restrict__ sq,
                       const float* __restrict__ sk, const float* __restrict__ temp,
                       float* __restrict__ oh) {
  const int it   = blockIdx.x & 15;
  const int bh   = blockIdx.x >> 4;     // b*3 + head
  const int head = bh % NH;
  const int row0 = it * 16;
  const int R0   = bh * HH;

  __shared__ float    S[16][257];
  __shared__ _Float16 P[16][256];
  __shared__ uint32_t VtP[16][516];     // [pair p][col]: {V[2p][n], V[2p+1][n]}
  __shared__ float    rq[256], rk[256];
  __shared__ float    red[16][17];

  const int t    = threadIdx.x;
  const int wave = t >> 5;
  const int lane = t & 31;
  const int lm   = lane & 15;
  const int hi   = lane >> 4;
  const float tv = temp[head];

  {
    float a0 = sq[R0 + t], c0 = sk[R0 + t];
    rq[t] = 1.f / fmaxf(sqrtf(a0), 1e-12f);
    rk[t] = 1.f / fmaxf(sqrtf(c0), 1e-12f);
  }
  __syncthreads();

  // ---- Phase 1: S = Q K^T --------------------------------------------------
  {
    const uint32_t* aptr  = (const uint32_t*)(qh + (size_t)(R0 + row0 + lm) * FF);
    const int j0a = wave * 32, j0b = wave * 32 + 16;
    const uint32_t* bptrA = (const uint32_t*)(kh + (size_t)(R0 + j0a + lm) * FF);
    const uint32_t* bptrB = (const uint32_t*)(kh + (size_t)(R0 + j0b + lm) * FF);
    v8f acc0 = {}, acc1 = {};
    for (int fu = 0; fu < FF / 2; fu += 16) {          // 32 halfs (16 dwords) / step
      AU a, b0, b1;
      #pragma unroll
      for (int i = 0; i < 4; ++i) {                    // K = 8*hi + 2i (,+1)
        int off = fu + 4 * hi + i;
        a.u[i] = aptr[off]; b0.u[i] = bptrA[off]; b1.u[i] = bptrB[off];
      }
      #pragma unroll
      for (int i = 4; i < 8; ++i) {                    // K = 16 + 8*hi + 2(i-4)
        int off = fu + 8 + 4 * hi + (i - 4);
        a.u[i] = aptr[off]; b0.u[i] = bptrA[off]; b1.u[i] = bptrB[off];
      }
      acc0 = __builtin_amdgcn_wmma_f32_16x16x32_f16(false, a.v, false, b0.v,
                                                    (short)0, acc0, false, false);
      acc1 = __builtin_amdgcn_wmma_f32_16x16x32_f16(false, a.v, false, b1.v,
                                                    (short)0, acc1, false, false);
    }
    #pragma unroll
    for (int r = 0; r < 8; ++r) {                      // D: lane=(n + 16*hi), M=r+8*hi
      S[r + 8 * hi][j0a + lm] = acc0[r];
      S[r + 8 * hi][j0b + lm] = acc1[r];
    }
  }
  __syncthreads();

  // ---- Phase 2: scale + softmax -------------------------------------------
  {
    const int r = t >> 4, sub = t & 15;                // 16 threads per row
    const float rqv = rq[row0 + r];
    float m = -3.0e38f;
    #pragma unroll
    for (int k2 = 0; k2 < 16; ++k2) {
      int j = sub + 16 * k2;
      float v = S[r][j] * (rqv * rk[j] * tv);
      S[r][j] = v;
      m = fmaxf(m, v);
    }
    red[r][sub] = m;
    __syncthreads();
    float mrow = -3.0e38f;
    #pragma unroll
    for (int k2 = 0; k2 < 16; ++k2) mrow = fmaxf(mrow, red[r][k2]);
    __syncthreads();
    float s = 0.f;
    #pragma unroll
    for (int k2 = 0; k2 < 16; ++k2) {
      int j = sub + 16 * k2;
      float e = __expf(S[r][j] - mrow);
      S[r][j] = e;
      s += e;
    }
    red[r][sub] = s;
    __syncthreads();
    float srow = 0.f;
    #pragma unroll
    for (int k2 = 0; k2 < 16; ++k2) srow += red[r][k2];
    const float inv = 1.f / srow;
    #pragma unroll
    for (int k2 = 0; k2 < 16; ++k2) {
      int j = sub + 16 * k2;
      P[r][j] = (_Float16)(S[r][j] * inv);
    }
  }
  __syncthreads();

  // ---- Phase 3: O = P V ----------------------------------------------------
  v8f vzero = {};
  for (int nb = 0; nb < 8; ++nb) {                     // 8 blocks of 512 f-cols
    const int n0blk = nb * 512;
    v8f acc[4] = {vzero, vzero, vzero, vzero};
    for (int j0 = 0; j0 < 256; j0 += 32) {
      __syncthreads();
      // Cooperative pair-interleaved stage of V[j0..j0+31][n0blk..n0blk+511].
      // Coalesced 16B global reads; ds_store_b128 writes.
      for (int idx = t; idx < 16 * 64; idx += 256) {   // 4 iters/thread
        const int p = idx >> 6;
        const int n = (idx & 63) * 8;
        const uint4 va = *(const uint4*)&vh[(size_t)(R0 + j0 + 2 * p    ) * FF + n0blk + n];
        const uint4 vb4= *(const uint4*)&vh[(size_t)(R0 + j0 + 2 * p + 1) * FF + n0blk + n];
        uint4 o01, o23;
        o01.x = (va.x & 0xffffu) | (vb4.x << 16);
        o01.y = (va.x >> 16)     | (vb4.x & 0xffff0000u);
        o01.z = (va.y & 0xffffu) | (vb4.y << 16);
        o01.w = (va.y >> 16)     | (vb4.y & 0xffff0000u);
        o23.x = (va.z & 0xffffu) | (vb4.z << 16);
        o23.y = (va.z >> 16)     | (vb4.z & 0xffff0000u);
        o23.z = (va.w & 0xffffu) | (vb4.w << 16);
        o23.w = (va.w >> 16)     | (vb4.w & 0xffff0000u);
        uint4* dst = (uint4*)&VtP[p][n];
        dst[0] = o01;
        dst[1] = o23;
      }
      __syncthreads();
      AU pa;                                           // A = P[16][32] from LDS
      #pragma unroll
      for (int i = 0; i < 8; ++i) {
        int off = (i < 4) ? (8 * hi + 2 * i) : (16 + 8 * hi + 2 * (i - 4));
        pa.u[i] = *(const uint32_t*)&P[lm][j0 + off];
      }
      #pragma unroll
      for (int tt = 0; tt < 4; ++tt) {                 // 4 n-tiles per wave
        const int col = (wave * 4 + tt) * 16 + lm;
        AU vb;                                         // B pairs: one b32 per VGPR
        #pragma unroll
        for (int i = 0; i < 8; ++i) {
          int p_idx = (i < 4) ? (4 * hi + i) : (8 + 4 * hi + (i - 4));
          vb.u[i] = VtP[p_idx][col];
        }
        acc[tt] = __builtin_amdgcn_wmma_f32_16x16x32_f16(false, pa.v, false, vb.v,
                                                         (short)0, acc[tt], false, false);
      }
    }
    #pragma unroll
    for (int tt = 0; tt < 4; ++tt) {
      const int f = n0blk + (wave * 4 + tt) * 16 + lm;
      #pragma unroll
      for (int r = 0; r < 8; ++r)
        oh[(size_t)(R0 + row0 + r + 8 * hi) * FF + f] = acc[tt][r];
    }
  }
}

// ---------------------------------------------------------------------------
// K4: conv1x1 48 -> 48 reading head layout, writing NCHW f32 output.
// One block = one (b, h) row of 256 pixels.
// ---------------------------------------------------------------------------
__global__ void k_conv_out(const float* __restrict__ oh, const float* __restrict__ wo,
                           float* __restrict__ out) {
  const int h = blockIdx.x & 255;
  const int b = blockIdx.x >> 8;
  __shared__ float xs[48][257];
  for (int i = threadIdx.x; i < 48 * 256; i += 256) {
    int c = i >> 8, w = i & 255;
    xs[c][w] = oh[(size_t)((b * NH + (c >> 4)) * HH + h) * FF + (c & 15) * WW + w];
  }
  __syncthreads();
  const int w = threadIdx.x;
  for (int o = 0; o < CDIM; ++o) {
    float acc = 0.f;
    const float* wrow = wo + o * CDIM;
    #pragma unroll 8
    for (int c = 0; c < CDIM; ++c) acc += wrow[c] * xs[c][w];
    out[(size_t)((b * CDIM + o) * HH + h) * WW + w] = acc;
  }
}

// ---------------------------------------------------------------------------
extern "C" void kernel_launch(void* const* d_in, const int* in_sizes, int n_in,
                              void* d_out, int out_size, void* d_ws, size_t ws_size,
                              hipStream_t stream) {
  const float* x   = (const float*)d_in[0];   // [4,48,256,256]
  const float* wq  = (const float*)d_in[1];   // [144,48,1,1]
  const float* wd  = (const float*)d_in[2];   // [144,1,3,3]
  const float* tmp = (const float*)d_in[3];   // [3,1,1]
  const float* wo  = (const float*)d_in[4];   // [48,48,1,1]
  float* out = (float*)d_out;

  char* ws = (char*)d_ws;
  const size_t S_pre = (size_t)BN * C3 * HW * sizeof(_Float16);      // 75.5 MB
  const size_t S_qkv = (size_t)BN * NH * HH * FF * sizeof(_Float16); // 25.2 MB each

  _Float16* pre = (_Float16*)ws;               // dead after k_dw
  float*    oh  = (float*)ws;                  // f32 O-heads alias pre (50.3 MB)
  _Float16* qh  = (_Float16*)(ws + S_pre);
  _Float16* kh  = (_Float16*)(ws + S_pre + S_qkv);
  _Float16* vh  = (_Float16*)(ws + S_pre + 2 * S_qkv);
  float*    sq  = (float*)(ws + S_pre + 3 * S_qkv);
  float*    sk  = sq + 4096;

  k_zero    <<<(BN * NH * HH + 255) / 256, 256, 0, stream>>>(sq, sk);
  k_conv1   <<<BN * (HW / 128),            256, 0, stream>>>(x, wq, pre);
  k_dw      <<<BN * C3 * HH,               256, 0, stream>>>(pre, wd, qh, kh, vh, sq, sk);
  k_attn    <<<BN * NH * (HH / 16),        256, 0, stream>>>(qh, kh, vh, sq, sk, tmp, oh);
  k_conv_out<<<BN * HH,                    256, 0, stream>>>(oh, wo, out);
}
